// GAT_17952963297695
// MI455X (gfx1250) — compile-verified, matrix operand order checked by reading, not run on hardware
//
#include <hip/hip_runtime.h>
#include <hip/hip_bf16.h>
#include <math.h>

typedef __attribute__((ext_vector_type(16))) _Float16 v16h;
typedef __attribute__((ext_vector_type(8)))  float    v8f;

#define NODES 50000
#define NEDGE 800000
#define ETOT  (NEDGE + NODES)
#define IN_C  128
#define HIDC  64
#define NHEAD 4
#define OUTC  40
#define SLOPE 0.2f

// ---------------- utility ----------------
__global__ void k_fill(float* __restrict__ p, float v, int n) {
  int i = blockIdx.x * blockDim.x + threadIdx.x;
  if (i < n) p[i] = v;
}

// float atomic max via monotone integer reinterpretation (native int atomics)
__device__ __forceinline__ void atomicMaxF(float* addr, float v) {
  if (v >= 0.f) atomicMax((int*)addr, __float_as_int(v));
  else          atomicMin((unsigned int*)addr, __float_as_uint(v));
}

__device__ __forceinline__ void edge_sd(const int* __restrict__ src,
                                        const int* __restrict__ dst,
                                        int e, int& s, int& d) {
  if (e < NEDGE) { s = src[e]; d = dst[e]; }
  else           { s = e - NEDGE; d = s; }   // self loops appended
}

// ---------------- W pre-pack into per-lane B-fragment order (f16) -------------
// Layout: Wp[v16h index] with index = (ct*KB + kb)*32 + lane, KB = K/32.
// Lane holds column ct*16 + (lane&15); halves [0..7] = K rows k0..k0+7,
// halves [8..15] = K rows k0+16..k0+23, where k0 = kb*32 + (lane>>4)*8.
// Columns >= M packed as zeros (removes all bounds checks from the GEMM loop).
__global__ void k_pack_w(const float* __restrict__ W, _Float16* __restrict__ Wp,
                         int K, int M, int NT) {
  const int KB = K >> 5;
  int idx = blockIdx.x * blockDim.x + threadIdx.x;   // over NT*KB*32
  if (idx >= NT * KB * 32) return;
  int lane = idx & 31;
  int rest = idx >> 5;
  int kb = rest % KB;
  int ct = rest / KB;
  int l16 = lane & 15, hf = lane >> 4;
  int col = ct * 16 + l16;
  int k0  = kb * 32 + hf * 8;
  _Float16* o = Wp + (size_t)idx * 16;
  #pragma unroll
  for (int i = 0; i < 8; ++i) {
    float lo = 0.f, hi = 0.f;
    if (col < M) {
      lo = W[(size_t)(k0 + i) * M + col];
      hi = W[(size_t)(k0 + 16 + i) * M + col];
    }
    o[i]     = (_Float16)lo;
    o[i + 8] = (_Float16)hi;
  }
}

// ---------------- GEMM: Out[rows,M] = X[rows,K] @ W[K,M], f16 WMMA, f32 acc ----
// One wave per 16x64 output tile (4 column tiles share one A fragment ->
// 4 independent WMMAs per K-step, no hazard nops, 4x less A traffic).
__global__ void k_gemm_wmma(const float* __restrict__ X, const _Float16* __restrict__ Wp,
                            float* __restrict__ Out, int K, int M) {
  const int lane = threadIdx.x & 31;
  const int hf   = lane >> 4;
  const int l16  = lane & 15;
  const int row0 = blockIdx.y * 16;
  const int ct0  = blockIdx.x * 4;          // first of 4 column tiles
  const int KB   = K >> 5;

  v8f acc[4] = {};
  const float* arow = X + (size_t)(row0 + l16) * K;
  const v16h*  bbase = (const v16h*)Wp + (size_t)ct0 * KB * 32 + lane;

  for (int kb = 0; kb < KB; ++kb) {
    const float* ap = arow + kb * 32 + hf * 8;
    float4 a0 = *(const float4*)(ap);
    float4 a1 = *(const float4*)(ap + 4);
    float4 a2 = *(const float4*)(ap + 16);
    float4 a3 = *(const float4*)(ap + 20);
    v16h a;
    a[0]=(_Float16)a0.x;  a[1]=(_Float16)a0.y;  a[2]=(_Float16)a0.z;  a[3]=(_Float16)a0.w;
    a[4]=(_Float16)a1.x;  a[5]=(_Float16)a1.y;  a[6]=(_Float16)a1.z;  a[7]=(_Float16)a1.w;
    a[8]=(_Float16)a2.x;  a[9]=(_Float16)a2.y;  a[10]=(_Float16)a2.z; a[11]=(_Float16)a2.w;
    a[12]=(_Float16)a3.x; a[13]=(_Float16)a3.y; a[14]=(_Float16)a3.z; a[15]=(_Float16)a3.w;

    const v16h* bp = bbase + (size_t)kb * 32;
    v16h b0 = bp[0];
    v16h b1 = bp[(size_t)KB * 32];
    v16h b2 = bp[(size_t)2 * KB * 32];
    v16h b3 = bp[(size_t)3 * KB * 32];

    acc[0] = __builtin_amdgcn_wmma_f32_16x16x32_f16(false, a, false, b0, (short)0, acc[0], false, false);
    acc[1] = __builtin_amdgcn_wmma_f32_16x16x32_f16(false, a, false, b1, (short)0, acc[1], false, false);
    acc[2] = __builtin_amdgcn_wmma_f32_16x16x32_f16(false, a, false, b2, (short)0, acc[2], false, false);
    acc[3] = __builtin_amdgcn_wmma_f32_16x16x32_f16(false, a, false, b3, (short)0, acc[3], false, false);
  }

  #pragma unroll
  for (int t = 0; t < 4; ++t) {
    int col = (ct0 + t) * 16 + l16;
    if (col < M) {
      #pragma unroll
      for (int r = 0; r < 8; ++r)
        Out[(size_t)(row0 + hf * 8 + r) * M + col] = acc[t][r];
    }
  }
}

// ---------------- per-node attention logits: als/ald[n,h] = <h[n,h,:], a[h,:]> ----
__global__ void k_logits(const float* __restrict__ h, const float* __restrict__ a_s,
                         const float* __restrict__ a_d, float* __restrict__ als,
                         float* __restrict__ ald, int n, int H, int C) {
  int idx = blockIdx.x * blockDim.x + threadIdx.x;   // over n*H
  if (idx >= n * H) return;
  int hh = idx % H;
  const float* hp = h + (size_t)idx * C;
  float s = 0.f, d = 0.f;
  for (int c = 0; c < C; ++c) {
    float v = hp[c];
    s += v * a_s[hh * C + c];
    d += v * a_d[hh * C + c];
  }
  als[idx] = s;
  ald[idx] = d;
}

// ---------------- edge pass 1: segment max of leaky-relu logits ----------------
__global__ void k_edge_max(const int* __restrict__ src, const int* __restrict__ dst,
                           const float* __restrict__ als, const float* __restrict__ ald,
                           float* __restrict__ mmax, int H) {
  int idx = blockIdx.x * blockDim.x + threadIdx.x;   // over ETOT*H
  if (idx >= ETOT * H) return;
  int e = idx / H, hh = idx - e * H;
  int s, d; edge_sd(src, dst, e, s, d);
  float a = als[s * H + hh] + ald[d * H + hh];
  a = (a > 0.f) ? a : a * SLOPE;
  atomicMaxF(&mmax[d * H + hh], a);
}

// ---------------- edge pass 2: e = exp(a - m[dst]); denom += e; stash e -------
__global__ void k_edge_expsum(const int* __restrict__ src, const int* __restrict__ dst,
                              const float* __restrict__ als, const float* __restrict__ ald,
                              const float* __restrict__ mmax, float* __restrict__ den,
                              float* __restrict__ attn, int H) {
  int idx = blockIdx.x * blockDim.x + threadIdx.x;
  if (idx >= ETOT * H) return;
  int e = idx / H, hh = idx - e * H;
  int s, d; edge_sd(src, dst, e, s, d);
  float a = als[s * H + hh] + ald[d * H + hh];
  a = (a > 0.f) ? a : a * SLOPE;
  float w = __expf(a - mmax[d * H + hh]);
  attn[idx] = w;
  unsafeAtomicAdd(&den[d * H + hh], w);
}

// ---------------- edge pass 3: agg[dst,:] += attn * h[src,:] (float4 gather) ---
__global__ void k_aggregate(const int* __restrict__ src, const int* __restrict__ dst,
                            const float* __restrict__ h, const float* __restrict__ attn,
                            const float* __restrict__ den, float* __restrict__ agg,
                            int H, int C) {
  const int D = H * C;
  const int Q = D >> 2;                               // D divisible by 4 (256, 40)
  long long idx = (long long)blockIdx.x * blockDim.x + threadIdx.x;  // over ETOT*Q
  if (idx >= (long long)ETOT * Q) return;
  int e  = (int)(idx / Q);
  int j  = (int)(idx - (long long)e * Q) << 2;
  int hh = j / C;                                     // group of 4 stays in one head
  int s, d; edge_sd(src, dst, e, s, d);
  float w = attn[(size_t)e * H + hh] / den[d * H + hh];
  float4 hv = *(const float4*)(h + (size_t)s * D + j);
  float* op = agg + (size_t)d * D + j;
  unsafeAtomicAdd(op + 0, hv.x * w);
  unsafeAtomicAdd(op + 1, hv.y * w);
  unsafeAtomicAdd(op + 2, hv.z * w);
  unsafeAtomicAdd(op + 3, hv.w * w);
}

// ---------------- epilogues ----------------
__global__ void k_bias_relu(const float* __restrict__ agg, const float* __restrict__ b,
                            float* __restrict__ act, int n, int D) {
  int idx = blockIdx.x * blockDim.x + threadIdx.x;    // over n*D (12.8M)
  if (idx >= n * D) return;
  float v = agg[idx] + b[idx % D];
  act[idx] = v > 0.f ? v : 0.f;
}

__global__ void k_bias_logsoftmax(const float* __restrict__ agg, const float* __restrict__ b,
                                  float* __restrict__ out, int n) {
  int i = blockIdx.x * blockDim.x + threadIdx.x;
  if (i >= n) return;
  float v[OUTC];
  float mx = -3.4e38f;
  for (int c = 0; c < OUTC; ++c) {
    float t = agg[(size_t)i * OUTC + c] + b[c];
    v[c] = t;
    mx = fmaxf(mx, t);
  }
  float ssum = 0.f;
  for (int c = 0; c < OUTC; ++c) ssum += __expf(v[c] - mx);
  float l = __logf(ssum) + mx;
  for (int c = 0; c < OUTC; ++c) out[(size_t)i * OUTC + c] = v[c] - l;
}

// ---------------- orchestration ----------------
static inline int cdiv(long long a, long long b) { return (int)((a + b - 1) / b); }

extern "C" void kernel_launch(void* const* d_in, const int* in_sizes, int n_in,
                              void* d_out, int out_size, void* d_ws, size_t ws_size,
                              hipStream_t stream) {
  (void)in_sizes; (void)n_in; (void)out_size; (void)ws_size;
  const float* x   = (const float*)d_in[0];
  const int*   ei  = (const int*)d_in[1];
  const int*   src = ei;             // edge_index[0]
  const int*   dst = ei + NEDGE;     // edge_index[1]
  const float* W1  = (const float*)d_in[2];
  const float* as1 = (const float*)d_in[3];
  const float* ad1 = (const float*)d_in[4];
  const float* b1  = (const float*)d_in[5];
  const float* W2  = (const float*)d_in[6];
  const float* as2 = (const float*)d_in[7];
  const float* ad2 = (const float*)d_in[8];
  const float* b2  = (const float*)d_in[9];
  const float* W3  = (const float*)d_in[10];
  const float* as3 = (const float*)d_in[11];
  const float* ad3 = (const float*)d_in[12];
  const float* b3  = (const float*)d_in[13];
  float* out = (float*)d_out;

  // workspace carve (floats): 3*N*256 + 4*N*4 + ETOT*4 + 32K  ~= 170 MB
  float* ws   = (float*)d_ws;
  float* hbuf = ws;                                  // [N,256] GEMM out
  float* act  = hbuf + (size_t)NODES * 256;          // [N,256] layer input (post relu)
  float* agg  = act  + (size_t)NODES * 256;          // [N,256] aggregation accumulator
  float* als  = agg  + (size_t)NODES * 256;          // [N,H]
  float* ald  = als  + (size_t)NODES * NHEAD;
  float* mmax = ald  + (size_t)NODES * NHEAD;
  float* den  = mmax + (size_t)NODES * NHEAD;
  float* attn = den  + (size_t)NODES * NHEAD;        // [ETOT,H]
  _Float16* wpack = (_Float16*)(attn + (size_t)ETOT * NHEAD);  // <= 64K halfs, 32B aligned

  auto run_layer = [&](const float* in, const float* W, const float* a_s,
                       const float* a_d, int K, int H, int C) {
    const int M  = H * C;
    const int NT = cdiv(M, 64) * 4;                  // column tiles, padded to groups of 4
    const int KB = K >> 5;

    k_pack_w<<<cdiv((long long)NT * KB * 32, 256), 256, 0, stream>>>(W, wpack, K, M, NT);

    dim3 grid(NT / 4, NODES / 16);
    k_gemm_wmma<<<grid, 32, 0, stream>>>(in, wpack, hbuf, K, M);

    const int nh = NODES * H;
    k_logits<<<cdiv(nh, 256), 256, 0, stream>>>(hbuf, a_s, a_d, als, ald, NODES, H, C);
    k_fill<<<cdiv(nh, 256), 256, 0, stream>>>(mmax, -__builtin_huge_valf(), nh);
    k_fill<<<cdiv(nh, 256), 256, 0, stream>>>(den, 0.f, nh);
    k_fill<<<cdiv((long long)NODES * M, 256), 256, 0, stream>>>(agg, 0.f, NODES * M);

    const int eh = ETOT * H;
    k_edge_max<<<cdiv(eh, 256), 256, 0, stream>>>(src, dst, als, ald, mmax, H);
    k_edge_expsum<<<cdiv(eh, 256), 256, 0, stream>>>(src, dst, als, ald, mmax, den, attn, H);

    const long long aggT = (long long)ETOT * (M / 4);
    k_aggregate<<<cdiv(aggT, 256), 256, 0, stream>>>(src, dst, hbuf, attn, den, agg, H, C);
  };

  // layer 1: 128 -> 4x64, relu
  run_layer(x, W1, as1, ad1, IN_C, NHEAD, HIDC);
  k_bias_relu<<<cdiv((long long)NODES * 256, 256), 256, 0, stream>>>(agg, b1, act, NODES, 256);
  // layer 2: 256 -> 4x64, relu
  run_layer(act, W2, as2, ad2, NHEAD * HIDC, NHEAD, HIDC);
  k_bias_relu<<<cdiv((long long)NODES * 256, 256), 256, 0, stream>>>(agg, b2, act, NODES, 256);
  // layer 3: 256 -> 1x40, log_softmax
  run_layer(act, W3, as3, ad3, NHEAD * HIDC, 1, OUTC);
  k_bias_logsoftmax<<<cdiv(NODES, 128), 128, 0, stream>>>(agg, b3, out, NODES);
}
